// Decoder_31198642438495
// MI455X (gfx1250) — matrix operand order repaired and checked by reading, not compiled, not run on hardware
//
#include <hip/hip_runtime.h>
#include <cfloat>

// ---------------- problem constants (from reference) ----------------
#define BB    2
#define QQ    4
#define HH    32
#define HK    8
#define HD    128
#define DD    4096            // H*HD
#define KVL   4096            // cached keys
#define S_TOT 4100            // KVL + QQ
#define ROWS8 8               // B*Q rows in every skinny GEMM
#define NTILE 257             // ceil(S_TOT/16)
#define R_KEEP 410            // max(min(S,128), S - int(S*0.9))

typedef __attribute__((ext_vector_type(16))) _Float16 v16h;
typedef __attribute__((ext_vector_type(8)))  float    v8f;

// =====================================================================
// K1/K6: skinny GEMM  O[8][N] = A[8][K] @ W[K][N]   (bandwidth bound on W)
// 1 thread = 1 output column, 8 f32 accumulators; A chunk staged in LDS.
// =====================================================================
__global__ void gemm8(const float* __restrict__ A,   // [8][K]
                      const float* __restrict__ W,   // [K][N]
                      float* __restrict__ O,         // [8][N]
                      int K, int N) {
  __shared__ float sA[8 * 1024];
  const int c = blockIdx.x * blockDim.x + threadIdx.x;
  float acc[8] = {0.f,0.f,0.f,0.f,0.f,0.f,0.f,0.f};

  for (int k0 = 0; k0 < K; k0 += 1024) {
    for (int i = threadIdx.x; i < 8 * 1024; i += blockDim.x)
      sA[i] = A[(i >> 10) * K + k0 + (i & 1023)];
    __syncthreads();
    if (c < N) {
      if (k0 + 1024 < K)   // gfx1250 global_prefetch_b8 for next W chunk
        __builtin_prefetch(&W[(size_t)(k0 + 1024) * N + c], 0, 1);
      #pragma unroll 4
      for (int j = 0; j < 1024; ++j) {
        const float w = W[(size_t)(k0 + j) * N + c];
        #pragma unroll
        for (int r = 0; r < 8; ++r) acc[r] += sA[r * 1024 + j] * w;
      }
    }
    __syncthreads();
  }
  if (c < N)
    #pragma unroll
    for (int r = 0; r < 8; ++r) O[(size_t)r * N + c] = acc[r];
}

// =====================================================================
// K2: RoPE queries & new keys (positions 4096..4099), repack layouts,
//     copy v_new into [B,HK,Q,HD].
// =====================================================================
__global__ void rope_pack(const float* __restrict__ q_raw,  // [8][4096]
                          const float* __restrict__ k_raw,  // [8][1024]
                          const float* __restrict__ v_raw,  // [8][1024]
                          float* __restrict__ qr,           // [B,H,Q,HD]
                          float* __restrict__ knr,          // [B,HK,Q,HD]
                          float* __restrict__ vnw) {        // [B,HK,Q,HD]
  const int QP = BB * HH * QQ * 64;   // 16384 rope pairs (q)
  const int KP = BB * HK * QQ * 64;   //  4096 rope pairs (k_new)
  const int VC = BB * HK * QQ * HD;   //  8192 copies     (v_new)
  int tid = blockIdx.x * blockDim.x + threadIdx.x;
  if (tid < QP) {
    int dd = tid & 63; int t = tid >> 6;
    int q = t & 3; t >>= 2; int h = t & 31; int b = t >> 5;
    float inv = __powf(10000.f, -(float)(2 * dd) / 128.f);
    float sn, cs; __sincosf((float)(KVL + q) * inv, &sn, &cs);
    const float* src = q_raw + (b * QQ + q) * DD + h * HD;
    float x0 = src[dd], x1 = src[dd + 64];
    float* dst = qr + (((b * HH + h) * QQ + q) * HD);
    dst[dd]      = x0 * cs - x1 * sn;
    dst[dd + 64] = x1 * cs + x0 * sn;
  } else if (tid < QP + KP) {
    int i = tid - QP;
    int dd = i & 63; int t = i >> 6;
    int q = t & 3; t >>= 2; int hk = t & 7; int b = t >> 3;
    float inv = __powf(10000.f, -(float)(2 * dd) / 128.f);
    float sn, cs; __sincosf((float)(KVL + q) * inv, &sn, &cs);
    const float* src = k_raw + (b * QQ + q) * (HK * HD) + hk * HD;
    float x0 = src[dd], x1 = src[dd + 64];
    float* dst = knr + (((b * HK + hk) * QQ + q) * HD);
    dst[dd]      = x0 * cs - x1 * sn;
    dst[dd + 64] = x1 * cs + x0 * sn;
  } else if (tid < QP + KP + VC) {
    int i = tid - QP - KP;
    int d = i & 127; int t = i >> 7;
    int q = t & 3; t >>= 2; int hk = t & 7; int b = t >> 3;
    vnw[((b * HK + hk) * QQ + q) * HD + d] =
        v_raw[(b * QQ + q) * (HK * HD) + hk * HD + d];
  }
}

// =====================================================================
// K3: draft scores via WMMA f32_16x16x32_f16.
//   All conditionals are wave-uniform (staging / in-place rope passes);
//   fragment build is branch-free ds_load_b128 + f32->f16 converts.
// =====================================================================
__global__ void draft_wmma(const float* __restrict__ k_cache, // [B,H,KV,HD]
                           const float* __restrict__ qr,      // [B,H,Q,HD] roped
                           const float* __restrict__ knr,     // [B,HK,Q,HD] roped
                           float* __restrict__ draft) {       // [B*H*Q][S_TOT]
  __shared__ __align__(16) float sKey[4][16 * HD];  // 4 waves x 16 keys x 128
  __shared__ float sInv[64];
  const int lane = threadIdx.x & 31;
  const int wid  = threadIdx.x >> 5;
  if (threadIdx.x < 64)
    sInv[threadIdx.x] = __powf(10000.f, -(float)(2 * threadIdx.x) / 128.f);
  __syncthreads();

  const int bh = blockIdx.x;           // b*32+h
  const int b = bh >> 5, h = bh & 31;
  const int m = lane & 15, hi = lane >> 4;

  // ---- A fragments (once per wave): lane m<4 holds roped q row m.
  //      Vectorized: two 8-float runs per 32-K chunk; rows m>=4 zeroed
  //      by a multiply mask (branch-free).
  v16h A[4];
  {
    const float  msk  = (m < 4) ? 1.f : 0.f;
    const float* qrow = qr + (size_t)(bh * QQ + (m < 4 ? m : 0)) * HD;
    #pragma unroll
    for (int kc = 0; kc < 4; ++kc) {
      const float* p0 = qrow + kc * 32 + hi * 16;       // K = kc*32 + hi*16 + 0..7
      const float* p1 = p0 + 8;                         // K = +8 .. +15
      float4 a0 = *(const float4*)(p0);
      float4 a1 = *(const float4*)(p0 + 4);
      float4 b0 = *(const float4*)(p1);
      float4 b1 = *(const float4*)(p1 + 4);
      A[kc][0]  = (_Float16)(a0.x * msk); A[kc][1]  = (_Float16)(a0.y * msk);
      A[kc][2]  = (_Float16)(a0.z * msk); A[kc][3]  = (_Float16)(a0.w * msk);
      A[kc][4]  = (_Float16)(a1.x * msk); A[kc][5]  = (_Float16)(a1.y * msk);
      A[kc][6]  = (_Float16)(a1.z * msk); A[kc][7]  = (_Float16)(a1.w * msk);
      A[kc][8]  = (_Float16)(b0.x * msk); A[kc][9]  = (_Float16)(b0.y * msk);
      A[kc][10] = (_Float16)(b0.z * msk); A[kc][11] = (_Float16)(b0.w * msk);
      A[kc][12] = (_Float16)(b1.x * msk); A[kc][13] = (_Float16)(b1.y * msk);
      A[kc][14] = (_Float16)(b1.z * msk); A[kc][15] = (_Float16)(b1.w * msk);
    }
  }

  const int t = blockIdx.y * 4 + wid;  // tile index (wave-uniform)
  if (t >= NTILE) return;
  const int s0 = t * 16;

  // ---- pass 1: stage 16 keys (raw cached / roped new / zero tail).
  //      s is wave-uniform inside the loop => cheap scalar branches.
  float* ks = sKey[wid];
  for (int j = 0; j < 16; ++j) {
    const int s = s0 + j;
    float4 v = make_float4(0.f, 0.f, 0.f, 0.f);
    if (s < S_TOT) {
      const float* src = (s < KVL)
          ? (k_cache + ((size_t)(b * HH + h) * KVL + s) * HD)
          : (knr + (size_t)((b * HK + (h >> 2)) * QQ + (s - KVL)) * HD);
      v = *(const float4*)(src + lane * 4);
    }
    *(float4*)(ks + j * HD + lane * 4) = v;
  }

  // ---- pass 2: RoPE cached keys in place (wave-uniform branch).
  //      Lane l owns dims 4l..4l+3; rotate-half partner block is (l^16)*4.
  //      Same-wave LDS: loads precede the store in program order => safe.
  {
    const float sgn = (lane < 16) ? -1.f : 1.f;
    const int   dB  = (lane & 15) * 4;      // d & 63 base for this lane
    for (int j = 0; j < 16; ++j) {
      const int s = s0 + j;
      if (s < KVL) {
        float4 raw = *(float4*)(ks + j * HD + lane * 4);
        float4 prt = *(float4*)(ks + j * HD + ((lane ^ 16) * 4));
        float4 o; float sn, cs;
        float fs = (float)s;
        __sincosf(fs * sInv[dB + 0], &sn, &cs); o.x = raw.x * cs + sgn * prt.x * sn;
        __sincosf(fs * sInv[dB + 1], &sn, &cs); o.y = raw.y * cs + sgn * prt.y * sn;
        __sincosf(fs * sInv[dB + 2], &sn, &cs); o.z = raw.z * cs + sgn * prt.z * sn;
        __sincosf(fs * sInv[dB + 3], &sn, &cs); o.w = raw.w * cs + sgn * prt.w * sn;
        *(float4*)(ks + j * HD + lane * 4) = o;
      }
    }
  }

  // ---- branch-free B fragments + 4 chained WMMAs.
  //      Lane holds column n = lane&15 (key s0+n), K = kc*32 + hi*16 + e.
  const int n = lane & 15;
  const int s = s0 + n;
  v8f c = {};
  #pragma unroll
  for (int kc = 0; kc < 4; ++kc) {
    const float* kp = ks + n * HD + kc * 32 + hi * 16;
    float4 x0 = *(const float4*)(kp);
    float4 x1 = *(const float4*)(kp + 4);
    float4 x2 = *(const float4*)(kp + 8);
    float4 x3 = *(const float4*)(kp + 12);
    v16h bb;
    bb[0]  = (_Float16)x0.x; bb[1]  = (_Float16)x0.y;
    bb[2]  = (_Float16)x0.z; bb[3]  = (_Float16)x0.w;
    bb[4]  = (_Float16)x1.x; bb[5]  = (_Float16)x1.y;
    bb[6]  = (_Float16)x1.z; bb[7]  = (_Float16)x1.w;
    bb[8]  = (_Float16)x2.x; bb[9]  = (_Float16)x2.y;
    bb[10] = (_Float16)x2.z; bb[11] = (_Float16)x2.w;
    bb[12] = (_Float16)x3.x; bb[13] = (_Float16)x3.y;
    bb[14] = (_Float16)x3.z; bb[15] = (_Float16)x3.w;
    c = __builtin_amdgcn_wmma_f32_16x16x32_f16(
            false, A[kc], false, bb, (short)0, c, false, false);
  }

  // ---- C layout: M = r + 8*hi, N = lane&15; valid q rows are M<4 ----
  if (hi == 0 && s < S_TOT) {
    #pragma unroll
    for (int r = 0; r < 4; ++r)
      draft[(size_t)(bh * QQ + r) * S_TOT + s] = c[r];
  }
}

// =====================================================================
// K4: exact R-th-largest threshold per row via 32-bit radix binary
//     search on sortable-uint mapping of f32. One block per row.
// =====================================================================
__global__ void topk_thresh(const float* __restrict__ draft,
                            unsigned* __restrict__ thr, int R) {
  __shared__ unsigned su[S_TOT];
  __shared__ int red[256];
  const int row = blockIdx.x;
  const float* dr = draft + (size_t)row * S_TOT;
  for (int s = threadIdx.x; s < S_TOT; s += 256) {
    unsigned bits = __float_as_uint(dr[s]);
    su[s] = (bits & 0x80000000u) ? ~bits : (bits | 0x80000000u);
  }
  __syncthreads();
  unsigned T = 0;
  for (int bit = 31; bit >= 0; --bit) {
    unsigned cand = T | (1u << bit);
    int local = 0;
    for (int s = threadIdx.x; s < S_TOT; s += 256) local += (su[s] >= cand);
    red[threadIdx.x] = local;
    __syncthreads();
    for (int off = 128; off > 0; off >>= 1) {
      if (threadIdx.x < off) red[threadIdx.x] += red[threadIdx.x + off];
      __syncthreads();
    }
    int total = red[0];
    __syncthreads();
    if (total >= R) T = cand;   // uniform across block
  }
  if (threadIdx.x == 0) thr[row] = T;
}

// =====================================================================
// K5: masked softmax + sparse attn@V. Only kept (~410) V rows are read:
//     ~10x less V bandwidth than dense. One block (128 thr) per row;
//     thread d owns output dim d.
// =====================================================================
#define CAPK 1024
__global__ void attn_out(const float* __restrict__ draft,
                         const unsigned* __restrict__ thr,
                         const float* __restrict__ v_cache,  // [B,H,KV,HD]
                         const float* __restrict__ vnw,      // [B,HK,Q,HD]
                         float* __restrict__ out_heads) {    // [8][4096]
  __shared__ int   ls[CAPK];
  __shared__ float lf[CAPK];
  __shared__ float red[128];
  __shared__ int cnt;
  const int row = blockIdx.x;                 // (b*32+h)*4+q
  const int q = row & 3, h = (row >> 2) & 31, b = row >> 7;
  const float* dr = draft + (size_t)row * S_TOT;
  const unsigned T = thr[row];
  if (threadIdx.x == 0) cnt = 0;
  __syncthreads();

  float lmax = -FLT_MAX;
  for (int s = threadIdx.x; s < S_TOT; s += 128) {
    float f = dr[s];
    unsigned bits = __float_as_uint(f);
    unsigned u = (bits & 0x80000000u) ? ~bits : (bits | 0x80000000u);
    if (u >= T) {
      int p = atomicAdd(&cnt, 1);
      if (p < CAPK) { ls[p] = s; lf[p] = f; }
      lmax = fmaxf(lmax, f);
    }
  }
  red[threadIdx.x] = lmax; __syncthreads();
  for (int off = 64; off > 0; off >>= 1) {
    if (threadIdx.x < off)
      red[threadIdx.x] = fmaxf(red[threadIdx.x], red[threadIdx.x + off]);
    __syncthreads();
  }
  const float mx = red[0];
  __syncthreads();

  const int n = min(cnt, CAPK);
  const float scale = 0.08838834764831845f;   // 1/sqrt(128)
  float lsum = 0.f;
  for (int i = threadIdx.x; i < n; i += 128) {
    float w = __expf((lf[i] - mx) * scale);
    lf[i] = w;
    lsum += w;
  }
  red[threadIdx.x] = lsum; __syncthreads();
  for (int off = 64; off > 0; off >>= 1) {
    if (threadIdx.x < off) red[threadIdx.x] += red[threadIdx.x + off];
    __syncthreads();
  }
  const float Z = red[0];
  __syncthreads();

  const int d = threadIdx.x;                  // 128 threads == HD
  float acc = 0.f;
  for (int i = 0; i < n; ++i) {
    int s = ls[i];
    float w = lf[i];
    const float* vr = (s < KVL)
        ? (v_cache + ((size_t)(b * HH + h) * KVL + s) * HD)
        : (vnw + (size_t)((b * HK + (h >> 2)) * QQ + (s - KVL)) * HD);
    acc += w * vr[d];
  }
  out_heads[(size_t)(b * QQ + q) * DD + h * HD + d] = acc / Z;
}

// =====================================================================
extern "C" void kernel_launch(void* const* d_in, const int* in_sizes, int n_in,
                              void* d_out, int out_size, void* d_ws, size_t ws_size,
                              hipStream_t stream) {
  const float* hidden  = (const float*)d_in[0];
  const float* k_cache = (const float*)d_in[1];
  const float* v_cache = (const float*)d_in[2];
  const float* Wq      = (const float*)d_in[3];
  const float* Wk      = (const float*)d_in[4];
  const float* Wv      = (const float*)d_in[5];
  const float* Wo      = (const float*)d_in[6];
  float* out = (float*)d_out;

  // ---- workspace layout (floats) ----
  float* ws     = (float*)d_ws;
  float* q_raw  = ws;                         // 8*4096
  float* k_raw  = q_raw + ROWS8 * DD;         // 8*1024
  float* v_raw  = k_raw + ROWS8 * HK * HD;    // 8*1024
  float* qr     = v_raw + ROWS8 * HK * HD;    // 32768
  float* knr    = qr + BB * HH * QQ * HD;     // 8192
  float* vnw    = knr + BB * HK * QQ * HD;    // 8192
  float* draft  = vnw + BB * HK * QQ * HD;    // 256*4100
  unsigned* thr = (unsigned*)(draft + (size_t)BB * HH * QQ * S_TOT); // 256
  float* out_h  = (float*)(thr + BB * HH * QQ);                      // 32768

  // 1) QKV projections (stream weights once)
  gemm8<<<dim3(DD / 256), 256, 0, stream>>>(hidden, Wq, q_raw, DD, DD);
  gemm8<<<dim3((HK * HD) / 256), 256, 0, stream>>>(hidden, Wk, k_raw, DD, HK * HD);
  gemm8<<<dim3((HK * HD) / 256), 256, 0, stream>>>(hidden, Wv, v_raw, DD, HK * HD);

  // 2) RoPE q / k_new, repack v_new
  rope_pack<<<dim3(112), 256, 0, stream>>>(q_raw, k_raw, v_raw, qr, knr, vnw);

  // 3) draft scores via WMMA (4 waves per block, 1 wave = 1 key tile)
  draft_wmma<<<dim3(BB * HH, (NTILE + 3) / 4), 128, 0, stream>>>(
      k_cache, qr, knr, draft);

  // 4) per-row top-k threshold
  topk_thresh<<<dim3(BB * HH * QQ), 256, 0, stream>>>(draft, thr, R_KEEP);

  // 5) masked softmax + sparse attn@V
  attn_out<<<dim3(BB * HH * QQ), 128, 0, stream>>>(draft, thr, v_cache, vnw, out_h);

  // 6) output projection
  gemm8<<<dim3(DD / 256), 256, 0, stream>>>(out_h, Wo, out, DD, DD);
}
